// NeuralParser_66915590472351
// MI455X (gfx1250) — matrix operand order repaired and checked by reading, not compiled, not run on hardware
//
#include <hip/hip_runtime.h>

// ---------------------------------------------------------------------------
// Neural dependency parser forward pass for MI455X (gfx1250, wave32).
//
//  * All dense GEMMs run through v_wmma_f32_16x16x32_f16 with f16-packed,
//    zero-padded operands: inner loop is pure b128 loads + 4x WMMA
//    (16x64 output strip per wave, A-fragment reused 4x).
//  * Char-LSTM recurrence: Whh as f16 fully resident in the 320KB WGP LDS
//    (320,000B weights + 4.8KB state, via dynamic LDS), b128 LDS reads.
//  * biLSTM recurrence: fwd+bwd in one kernel (gridDim=2, two WGPs in
//    parallel, each with its own gate buffer), Whh streamed as f16 from the
//    192MB L2 (1.28MB/step).
//  * Biaffine adj.T emitted by the same WMMA GEMM with non-temporal stores
//    (64MB output > L2; keep L2 for recurrence weights).
// ---------------------------------------------------------------------------

typedef __attribute__((ext_vector_type(16))) _Float16 v16h;
typedef __attribute__((ext_vector_type(8)))  _Float16 v8h;
typedef __attribute__((ext_vector_type(8)))  float    v8f;

#define NW      4096        // words
#define CPW     8
#define TCH     (NW*CPW)    // 32768 chars
#define UH      200
#define G4      (4*UH)      // 800
#define BH      400
#define GB      (4*BH)      // 1600
#define DW      100
#define DT      100

enum { ACT_NONE = 0, ACT_RELU = 1, ACT_LRELU = 2 };

// ---------------------------------------------------------------------------
// f32 [rows,K] -> f16 [rowsP,Kp], zero padded (Kp mult of 32, rowsP >= rows).
// ---------------------------------------------------------------------------
__global__ void pack_f16(const float* __restrict__ src, _Float16* __restrict__ dst,
                         int rows, int K, int rowsP, int Kp) {
    int i = blockIdx.x * blockDim.x + threadIdx.x;
    if (i >= rowsP * Kp) return;
    int r = i / Kp, c = i - r * Kp;
    float v = (r < rows && c < K) ? src[(size_t)r * K + c] : 0.f;
    dst[i] = (_Float16)v;
}

// flat f32 -> f16 convert (for recurrence weights)
__global__ void pack_flat_f16(const float* __restrict__ src,
                              _Float16* __restrict__ dst, int n) {
    int i = blockIdx.x * blockDim.x + threadIdx.x;
    if (i < n) dst[i] = (_Float16)src[i];
}

// ---------------------------------------------------------------------------
// WMMA GEMM on packed operands:
//   Y[N,M] = act(Xh[N,Kp] @ Wh[Mp,Kp]^T + bias[M])
// N mult of 16, Kp mult of 32, Mp mult of 64. One 16x64 strip per wave.
// Fragments use contiguous k = 16*half + s in A and B (A: two contiguous v8h chunks,
// B: one contiguous v16h chunk, C/D: VGPR v -> row v + half*8, col q).
// ---------------------------------------------------------------------------
__global__ void wmma_gemm_tiled(const _Float16* __restrict__ Xh,
                                const _Float16* __restrict__ Wh,
                                const float* __restrict__ bias,
                                float* __restrict__ Y,
                                int Nrows, int Kp, int M, int Mp, int act, int nt) {
    const int lane  = threadIdx.x & 31;
    const int half  = lane >> 4;
    const int q     = lane & 15;
    const int tilesM = Mp >> 6;                       // 64-wide strips
    const int tile   = blockIdx.x * 8 + (threadIdx.x >> 5);
    if (tile >= (Nrows >> 4) * tilesM) return;        // wave-uniform exit

    const int tn = tile / tilesM;
    const int tm = tile - tn * tilesM;
    const int n0 = tn << 4;
    const int m0 = tm << 6;

    const _Float16* xrow = Xh + (size_t)(n0 + q) * Kp + half * 16;
    const _Float16* wr   = Wh + (size_t)(m0 + q) * Kp + half * 16;
    const size_t wstride = (size_t)16 * Kp;

    v8f acc[4];
#pragma unroll
    for (int j = 0; j < 4; ++j) acc[j] = (v8f){0.f,0.f,0.f,0.f,0.f,0.f,0.f,0.f};

    for (int kb = 0; kb < Kp; kb += 32) {
        __builtin_prefetch(xrow + kb + 32, 0, 1);     // global_prefetch_b8
        __builtin_prefetch(wr + kb + 32, 0, 1);
        v8h alo = *(const v8h*)(xrow + kb);
        v8h ahi = *(const v8h*)(xrow + kb + 8);
        v16h a = __builtin_shufflevector(alo, ahi,
                    0,1,2,3,4,5,6,7,8,9,10,11,12,13,14,15);
#pragma unroll
        for (int j = 0; j < 4; ++j) {
            v16h b = *(const v16h*)(wr + (size_t)j * wstride + kb);
            acc[j] = __builtin_amdgcn_wmma_f32_16x16x32_f16(
                         false, a, false, b, (short)0, acc[j], false, false);
        }
    }

#pragma unroll
    for (int j = 0; j < 4; ++j) {
        int col = m0 + j * 16 + q;
        if (col < M) {
            const float bs = bias ? bias[col] : 0.f;
#pragma unroll
            for (int v = 0; v < 8; ++v) {
                int row = n0 + v + half * 8;
                float y = acc[j][v] + bs;
                if (act == ACT_RELU)       y = fmaxf(y, 0.f);
                else if (act == ACT_LRELU) y = (y > 0.f) ? y : 0.01f * y;
                float* p = &Y[(size_t)row * M + col];
                if (nt) __builtin_nontemporal_store(y, p);   // store TH=NT
                else    *p = y;
            }
        }
    }
}

// ---------------------------------------------------------------------------
// Elementwise helpers
// ---------------------------------------------------------------------------
__global__ void gather_rows(const float* __restrict__ emb,
                            const int* __restrict__ ids,
                            float* __restrict__ out, int n, int d) {
    int i = blockIdx.x * blockDim.x + threadIdx.x;
    if (i >= n * d) return;
    int r = i / d, cidx = i - r * d;
    out[i] = emb[(size_t)ids[r] * d + cidx];
}

__global__ void build_x0(const float* __restrict__ word_emb,
                         const int* __restrict__ words,
                         const float* __restrict__ h_char,
                         const float* __restrict__ tag_emb,
                         const int* __restrict__ tags,
                         float* __restrict__ x) {
    int i = blockIdx.x * blockDim.x + threadIdx.x;
    if (i >= NW * (DW + DT)) return;
    int r = i / (DW + DT), cidx = i - r * (DW + DT);
    if (cidx < DW)
        x[i] = word_emb[(size_t)words[r] * DW + cidx] + h_char[r * DW + cidx];
    else
        x[i] = tag_emb[(size_t)tags[r] * DT + (cidx - DW)];
}

__global__ void concat2(const float* __restrict__ a, const float* __restrict__ b,
                        float* __restrict__ out, int n, int d) {
    int i = blockIdx.x * blockDim.x + threadIdx.x;
    if (i >= n * 2 * d) return;
    int r = i / (2 * d), cidx = i - r * 2 * d;
    out[i] = (cidx < d) ? a[r * d + cidx] : b[r * d + (cidx - d)];
}

__device__ __forceinline__ float sigmf(float x) { return 1.f / (1.f + __expf(-x)); }

// ---------------------------------------------------------------------------
// Char LSTM recurrence (torch gate order i,f,g,o), 1 workgroup x 1024 thr.
// Whh lives in LDS as f16: 800*200*2 = 320,000B (+4.8KB state) -> fits the
// 320KB WGP LDS. Dynamic LDS so no static limit applies.
// ---------------------------------------------------------------------------
#define CHAR_LDS_BYTES (320000 + 3200 + 800 + 800)   // W, g, h, c = 324,800B

__global__ void lstm_char(const float* __restrict__ Xg,   // [TCH, 800]
                          const float* __restrict__ Whh,  // [800, 200] f32
                          float* __restrict__ hs) {       // [TCH, 200]
    extern __shared__ char smem[];
    _Float16* W = (_Float16*)smem;                     // [800*200]
    float* g    = (float*)(smem + 320000);             // [800]
    float* h    = (float*)(smem + 323200);             // [200]
    float* cs   = (float*)(smem + 324000);             // [200]
    const int tid = threadIdx.x;
    for (int i = tid; i < G4 * UH; i += 1024) W[i] = (_Float16)Whh[i];
    if (tid < UH) { h[tid] = 0.f; cs[tid] = 0.f; }
    __syncthreads();
    const float4* h4 = (const float4*)h;
    for (int t = 0; t < TCH; ++t) {
        const float* xg = Xg + (size_t)t * G4;
        if (tid < G4) {
            const v8h* wr = (const v8h*)(W + tid * UH);   // 400B stride, 16B aligned
            float acc = xg[tid];
#pragma unroll
            for (int k8 = 0; k8 < UH / 8; ++k8) {         // 25x: ds_load_b128
                v8h wv = wr[k8];
                float4 h0 = h4[2 * k8], h1 = h4[2 * k8 + 1];
                acc = __builtin_fmaf((float)wv[0], h0.x, acc);
                acc = __builtin_fmaf((float)wv[1], h0.y, acc);
                acc = __builtin_fmaf((float)wv[2], h0.z, acc);
                acc = __builtin_fmaf((float)wv[3], h0.w, acc);
                acc = __builtin_fmaf((float)wv[4], h1.x, acc);
                acc = __builtin_fmaf((float)wv[5], h1.y, acc);
                acc = __builtin_fmaf((float)wv[6], h1.z, acc);
                acc = __builtin_fmaf((float)wv[7], h1.w, acc);
            }
            g[tid] = acc;
        }
        __syncthreads();
        if (tid < UH) {
            float ig = sigmf(g[tid]);
            float fg = sigmf(g[UH + tid]);
            float gg = tanhf(g[2 * UH + tid]);
            float og = sigmf(g[3 * UH + tid]);
            float cn = fg * cs[tid] + ig * gg;
            cs[tid] = cn;
            float hn = og * tanhf(cn);
            h[tid] = hn;
            hs[(size_t)t * UH + tid] = hn;
        }
        __syncthreads();
    }
}

// ---------------------------------------------------------------------------
// biLSTM recurrence: both directions concurrently (blockIdx.x = dir, two
// WGPs), each direction reading its own gate buffer. Whh pre-packed f16
// [2,1600,400] streamed from L2 (1.28MB/step).
// ---------------------------------------------------------------------------
__global__ void lstm_bi2(const float* __restrict__ Xg0,    // [NW, 1600] dir0
                         const float* __restrict__ Xg1,    // [NW, 1600] dir1
                         const _Float16* __restrict__ WhhH,// [2, 1600, 400]
                         float* __restrict__ fwd,          // [NW, 400]
                         float* __restrict__ bwd) {        // [NW, 400]
    __shared__ float h[BH], cs[BH], g[GB];
    const int tid = threadIdx.x;
    const int dir = blockIdx.x;
    const float* Xg = dir ? Xg1 : Xg0;
    const _Float16* Wd = WhhH + (size_t)dir * GB * BH;
    float* out = dir ? bwd : fwd;
    for (int j = tid; j < BH; j += 1024) { h[j] = 0.f; cs[j] = 0.f; }
    __syncthreads();
    const float4* h4 = (const float4*)h;
    for (int s = 0; s < NW; ++s) {
        const int t = dir ? (NW - 1 - s) : s;
        const float* xg = Xg + (size_t)t * GB;
        for (int j = tid; j < GB; j += 1024) {
            const v8h* wr = (const v8h*)(Wd + (size_t)j * BH);  // 800B stride
            float acc = xg[j];
#pragma unroll 5
            for (int k8 = 0; k8 < BH / 8; ++k8) {               // 50x b128
                v8h wv = wr[k8];
                float4 h0 = h4[2 * k8], h1 = h4[2 * k8 + 1];
                acc = __builtin_fmaf((float)wv[0], h0.x, acc);
                acc = __builtin_fmaf((float)wv[1], h0.y, acc);
                acc = __builtin_fmaf((float)wv[2], h0.z, acc);
                acc = __builtin_fmaf((float)wv[3], h0.w, acc);
                acc = __builtin_fmaf((float)wv[4], h1.x, acc);
                acc = __builtin_fmaf((float)wv[5], h1.y, acc);
                acc = __builtin_fmaf((float)wv[6], h1.z, acc);
                acc = __builtin_fmaf((float)wv[7], h1.w, acc);
            }
            g[j] = acc;
        }
        __syncthreads();
        for (int j = tid; j < BH; j += 1024) {
            float ig = sigmf(g[j]);
            float fg = sigmf(g[BH + j]);
            float gg = tanhf(g[2 * BH + j]);
            float og = sigmf(g[3 * BH + j]);
            float cn = fg * cs[j] + ig * gg;
            cs[j] = cn;
            float hn = og * tanhf(cn);
            h[j] = hn;
            out[(size_t)t * BH + j] = hn;
        }
        __syncthreads();
    }
}

// P = HH @ U1, r = HH @ u2^T
__global__ void biaffine_prep(const float* __restrict__ HH,
                              const float* __restrict__ U1,
                              const float* __restrict__ u2,
                              float* __restrict__ P, float* __restrict__ r) {
    int i = blockIdx.x * blockDim.x + threadIdx.x;
    if (i >= NW) return;
    float hh[20];
#pragma unroll
    for (int k = 0; k < 20; ++k) hh[k] = HH[i * 20 + k];
    float ar = 0.f;
#pragma unroll
    for (int k = 0; k < 20; ++k) ar = __builtin_fmaf(hh[k], u2[k], ar);
    r[i] = ar;
    for (int j = 0; j < 20; ++j) {
        float a = 0.f;
#pragma unroll
        for (int k = 0; k < 20; ++k) a = __builtin_fmaf(hh[k], U1[k * 20 + j], a);
        P[i * 20 + j] = a;
    }
}

// ---------------------------------------------------------------------------
// Host-side orchestration
// ---------------------------------------------------------------------------
static void gemm(hipStream_t s, const float* X, const float* W, const float* b,
                 float* Y, _Float16* XH, _Float16* WH,
                 int N, int K, int M, int act, int nt = 0) {
    const int Kp = (K + 31) & ~31;
    const int Mp = (M + 63) & ~63;
    pack_f16<<<((size_t)N * Kp + 255) / 256, 256, 0, s>>>(X, XH, N, K, N, Kp);
    pack_f16<<<((size_t)Mp * Kp + 255) / 256, 256, 0, s>>>(W, WH, M, K, Mp, Kp);
    int tiles = (N / 16) * (Mp / 64);
    wmma_gemm_tiled<<<(tiles + 7) / 8, 256, 0, s>>>(XH, WH, b, Y, N, Kp, M, Mp, act, nt);
}

extern "C" void kernel_launch(void* const* d_in, const int* in_sizes, int n_in,
                              void* d_out, int out_size, void* d_ws, size_t ws_size,
                              hipStream_t stream) {
    (void)in_sizes; (void)n_in; (void)out_size; (void)ws_size;
    const int*   char_ids = (const int*)d_in[0];
    const int*   words    = (const int*)d_in[1];
    const int*   tags     = (const int*)d_in[2];
    const int*   wend     = (const int*)d_in[3];
    const float* word_emb = (const float*)d_in[4];
    const float* tag_emb  = (const float*)d_in[5];
    const float* char_emb = (const float*)d_in[6];
    const float* uni_Wih  = (const float*)d_in[7];
    const float* uni_Whh  = (const float*)d_in[8];
    const float* uni_b    = (const float*)d_in[9];
    const float* ch_W1 = (const float*)d_in[10]; const float* ch_b1 = (const float*)d_in[11];
    const float* ch_W2 = (const float*)d_in[12]; const float* ch_b2 = (const float*)d_in[13];
    const float* bWih0  = (const float*)d_in[14];
    const float* bWhh0  = (const float*)d_in[15];
    const float* bb0    = (const float*)d_in[16];
    const float* bWih12 = (const float*)d_in[17];
    const float* bWhh12 = (const float*)d_in[18];
    const float* bb12   = (const float*)d_in[19];
    const float* ah_W1 = (const float*)d_in[20]; const float* ah_b1 = (const float*)d_in[21];
    const float* ah_W2 = (const float*)d_in[22]; const float* ah_b2 = (const float*)d_in[23];
    const float* ad_W1 = (const float*)d_in[24]; const float* ad_b1 = (const float*)d_in[25];
    const float* ad_W2 = (const float*)d_in[26]; const float* ad_b2 = (const float*)d_in[27];
    const float* lh_W1 = (const float*)d_in[28]; const float* lh_b1 = (const float*)d_in[29];
    const float* lh_W2 = (const float*)d_in[30]; const float* lh_b2 = (const float*)d_in[31];
    const float* ld_W1 = (const float*)d_in[32]; const float* ld_b1 = (const float*)d_in[33];
    const float* ld_W2 = (const float*)d_in[34]; const float* ld_b2 = (const float*)d_in[35];
    const float* cl_W1 = (const float*)d_in[36]; const float* cl_b1 = (const float*)d_in[37];
    const float* cl_W2 = (const float*)d_in[38]; const float* cl_b2 = (const float*)d_in[39];
    const float* U1 = (const float*)d_in[40];
    const float* u2 = (const float*)d_in[41];

    float* adj_out   = (float*)d_out;                        // [4096,4096] = adj.T
    float* label_out = (float*)d_out + (size_t)NW * NW;      // [4096,40]

    // ---- workspace layout (floats); total ~169 MB --------------------------
    float* ws = (float*)d_ws;
    float* Xc    = ws;                         // [32768,200]  (later reused: hsC)
    float* hsC   = ws;                         // reuse of Xc after char GEMM
    float* XgC   = ws + 6553600;               // [32768,800]  char gates region
    // region reuse after char recurrence (all within XgC's 26,214,400 floats):
    float* Xg0   = XgC;                        // [4096,1600] dir0 gates
    float* Xg1   = XgC + 6553600;              // [4096,1600] dir1 gates
    float* fwd   = XgC + 13107200;             // [4096,400]
    float* bwd   = XgC + 14745600;             // [4096,400]
    float* hA    = XgC + 16384000;             // [4096,800]
    float* hB    = XgC + 19660800;             // [4096,800]
    float* T500  = XgC + 22937600;             // [4096,500]
    float* T200  = XgC + 24985600;             // [4096,200]  (ends 25,804,800)
    float* sm    = ws + 32768000;              // small arena
    float* Hend  = sm;                         // [4096,200]
    float* T1    = sm + 819200;                // [4096,400]
    float* Hchar = sm + 2457600;               // [4096,100]
    float* x0    = sm + 2867200;               // [4096,200]
    float* HH    = sm + 3686400;               // [4096,20]
    float* HD    = sm + 3768320;               // [4096,20]
    float* LH    = sm + 3850240;               // [4096,20]
    float* LD    = sm + 3932160;               // [4096,20]
    float* LCAT  = sm + 4014080;               // [4096,40]
    float* T60   = sm + 4177920;               // [4096,60]
    float* P     = sm + 4423680;               // [4096,20]
    float* rv    = sm + 4505600;               // [4096]
    // f16 staging (64B-aligned offsets)
    _Float16* XH     = (_Float16*)(ws + 37277696);   // 7,340,032 halfs
    _Float16* WH     = (_Float16*)(ws + 40947712);   // 1,280,000 halfs
    _Float16* WhhH16 = (_Float16*)(ws + 41587712);   // 1,280,000 halfs [2,1600,400]

    // ---- 1) char stream ----------------------------------------------------
    gather_rows<<<(TCH * UH + 255) / 256, 256, 0, stream>>>(char_emb, char_ids, Xc, TCH, UH);
    gemm(stream, Xc, uni_Wih, uni_b, XgC, XH, WH, TCH, UH, G4, ACT_NONE);
    lstm_char<<<1, 1024, CHAR_LDS_BYTES, stream>>>(XgC, uni_Whh, hsC);
    gather_rows<<<(NW * UH + 255) / 256, 256, 0, stream>>>(hsC, wend, Hend, NW, UH);
    gemm(stream, Hend, ch_W1, ch_b1, T1, XH, WH, NW, UH, 2 * UH, ACT_LRELU);
    gemm(stream, T1, ch_W2, ch_b2, Hchar, XH, WH, NW, 2 * UH, DW, ACT_NONE);
    build_x0<<<(NW * (DW + DT) + 255) / 256, 256, 0, stream>>>(word_emb, words, Hchar,
                                                               tag_emb, tags, x0);

    // ---- 2) 3-layer biLSTM (fwd+bwd concurrent per layer) ------------------
    const float* lin = x0;
    float* buf[2] = { hA, hB };
    for (int l = 0; l < 3; ++l) {
        const int Kin = (l == 0) ? (DW + DT) : (2 * BH);
        const float* WihBase = (l == 0) ? bWih0 : (bWih12 + (size_t)(l - 1) * 2 * GB * (2 * BH));
        const float* WhhBase = (l == 0) ? bWhh0 : (bWhh12 + (size_t)(l - 1) * 2 * GB * BH);
        const float* bbBase  = (l == 0) ? bb0   : (bb12   + (size_t)(l - 1) * 2 * GB);
        // pack both directions' Whh to f16 (contiguous [2,1600,400])
        pack_flat_f16<<<(2 * GB * BH + 255) / 256, 256, 0, stream>>>(WhhBase, WhhH16,
                                                                     2 * GB * BH);
        // per-direction input-gate GEMMs into disjoint buffers
        gemm(stream, lin, WihBase, bbBase, Xg0, XH, WH, NW, Kin, GB, ACT_NONE);
        gemm(stream, lin, WihBase + (size_t)GB * Kin, bbBase + GB, Xg1, XH, WH,
             NW, Kin, GB, ACT_NONE);
        // both directions run concurrently on two WGPs
        lstm_bi2<<<2, 1024, 0, stream>>>(Xg0, Xg1, WhhH16, fwd, bwd);
        float* hout = buf[l & 1];
        concat2<<<(NW * 2 * BH + 255) / 256, 256, 0, stream>>>(fwd, bwd, hout, NW, BH);
        lin = hout;
    }
    const float* o2 = lin;   // [4096, 800]

    // ---- 3) MLP heads ------------------------------------------------------
    gemm(stream, o2, ah_W1, ah_b1, T500, XH, WH, NW, 2 * BH, 500, ACT_RELU);
    gemm(stream, T500, ah_W2, ah_b2, HH, XH, WH, NW, 500, 20, ACT_NONE);
    gemm(stream, o2, ad_W1, ad_b1, T500, XH, WH, NW, 2 * BH, 500, ACT_RELU);
    gemm(stream, T500, ad_W2, ad_b2, HD, XH, WH, NW, 500, 20, ACT_NONE);
    gemm(stream, o2, lh_W1, lh_b1, T200, XH, WH, NW, 2 * BH, 200, ACT_RELU);
    gemm(stream, T200, lh_W2, lh_b2, LH, XH, WH, NW, 200, 20, ACT_NONE);
    gemm(stream, o2, ld_W1, ld_b1, T200, XH, WH, NW, 2 * BH, 200, ACT_RELU);
    gemm(stream, T200, ld_W2, ld_b2, LD, XH, WH, NW, 200, 20, ACT_NONE);

    // labels head
    concat2<<<(NW * 40 + 255) / 256, 256, 0, stream>>>(LH, LD, LCAT, NW, 20);
    gemm(stream, LCAT, cl_W1, cl_b1, T60, XH, WH, NW, 40, 60, ACT_RELU);
    gemm(stream, T60, cl_W2, cl_b2, label_out, XH, WH, NW, 60, 40, ACT_NONE);

    // ---- 4) biaffine (non-temporal 64MB store) -----------------------------
    biaffine_prep<<<(NW + 255) / 256, 256, 0, stream>>>(HH, U1, u2, P, rv);
    gemm(stream, HD, P, rv, adj_out, XH, WH, NW, 20, NW, ACT_NONE, /*nt=*/1);
}
